// LSTMPredictor_35794257445039
// MI455X (gfx1250) — compile-verified, matrix operand order checked by reading, not run on hardware
//
#include <hip/hip_runtime.h>
#include <hip/hip_bf16.h>

#define Bn 64
#define Tn 2048
#define Hn 256

typedef __attribute__((ext_vector_type(16))) __bf16 v16bf;
typedef __attribute__((ext_vector_type(8)))  __bf16 v8bf;
typedef __attribute__((ext_vector_type(8)))  float  v8f;

#define NBLOCKS  8
#define NTHREADS 256
#define WAVES_PER_BLOCK 8

// ---------------- helpers ----------------

// Fast activations on the hardware transcendental path (v_exp_f32 / v_rcp_f32).
__device__ __forceinline__ float sigmoidf_(float v) {
    return __builtin_amdgcn_rcpf(1.0f + __expf(-v));
}
__device__ __forceinline__ float tanhf_(float v) {
    // tanh(x) = 1 - 2/(1+exp(2x));  exp->inf => 1, exp->0 => -1 (correct limits)
    return 1.0f - 2.0f * __builtin_amdgcn_rcpf(1.0f + __expf(2.0f * v));
}

// Load a 16-element bf16 fragment for WMMA A/B operands.
// Per documented 16-bit 16x32 layout: each lane supplies two contiguous
// 8-element (16-byte) chunks at k and k+16 (lane>=16 pre-offset by +8).
__device__ __forceinline__ v16bf load_frag(const __bf16* p) {
    v8bf lo = *(const v8bf*)p;
    v8bf hi = *(const v8bf*)(p + 16);
    v16bf r;
#pragma unroll
    for (int i = 0; i < 8; ++i) { r[i] = lo[i]; r[i + 8] = hi[i]; }
    return r;
}

// Device-wide barrier for a small persistent grid (monotone counter).
__device__ __forceinline__ void grid_sync(unsigned* cnt, unsigned nb) {
    __syncthreads();
    if (threadIdx.x == 0) {
        __threadfence();                       // release our stores
        unsigned arrival = atomicAdd(cnt, 1u);
        unsigned gen = arrival / nb;
        while (atomicAdd(cnt, 0u) / nb <= gen) {
            __builtin_amdgcn_s_sleep(1);
        }
    }
    __syncthreads();
    __threadfence();                           // acquire peer stores
}

// ---------------- init: weight convert / state & output init ----------------

__global__ void lstm_init_kernel(const float* __restrict__ Whh0,
                                 const float* __restrict__ bih0, const float* __restrict__ bhh0,
                                 const float* __restrict__ Wih1, const float* __restrict__ Whh1,
                                 const float* __restrict__ bih1, const float* __restrict__ bhh1,
                                 const float* __restrict__ fc_b,
                                 float* __restrict__ out,
                                 __bf16* __restrict__ whh0b, __bf16* __restrict__ w1b,
                                 float* __restrict__ b0s, float* __restrict__ b1s,
                                 __bf16* __restrict__ h0buf, __bf16* __restrict__ h1buf,
                                 unsigned* __restrict__ bar) {
    const long tid    = (long)blockIdx.x * blockDim.x + threadIdx.x;
    const long stride = (long)gridDim.x * blockDim.x;

    // Concatenated layer-1 weights: rows [4H], K = [Wih1 (256) | Whh1 (256)]
    for (long k = tid; k < 1024L * 512; k += stride) {
        int g = (int)(k >> 9), kk = (int)(k & 511);
        float v = (kk < 256) ? Wih1[g * 256 + kk] : Whh1[g * 256 + (kk - 256)];
        w1b[k] = (__bf16)v;
    }
    for (long k = tid; k < 1024L * 256; k += stride) whh0b[k] = (__bf16)Whh0[k];
    for (long k = tid; k < 1024; k += stride) {
        b0s[k] = bih0[k] + bhh0[k];
        b1s[k] = bih1[k] + bhh1[k];
    }
    for (long k = tid; k < 2L * Bn * Hn; k += stride) {
        h0buf[k] = (__bf16)0.0f;
        h1buf[k] = (__bf16)0.0f;
    }
    const float fb = fc_b[0];
    for (long k = tid; k < (long)Bn * Tn; k += stride) out[k] = fb;
    if (tid == 0) *bar = 0u;
}

// ---------------- persistent fused 2-layer LSTM scan ----------------
// Software-pipelined: iteration t computes layer-0 at time t AND layer-1 at
// time t-1 (both only read state published at the previous sync), so a single
// device-wide barrier per iteration suffices (2048 syncs instead of 4096).

__global__ void __launch_bounds__(NTHREADS)
lstm_persistent_kernel(const float* __restrict__ x,      // [B,T,1]
                       const float* __restrict__ Wih0,   // [4H,1]
                       const float* __restrict__ fcw,    // [1,H]
                       const __bf16* __restrict__ whh0b, // [4H,H] bf16
                       const __bf16* __restrict__ w1b,   // [4H,2H] bf16 (Wih1|Whh1)
                       const float* __restrict__ b0s, const float* __restrict__ b1s,
                       __bf16* __restrict__ h0buf,       // 2 x [B,H] bf16
                       __bf16* __restrict__ h1buf,       // 2 x [B,H] bf16
                       float* __restrict__ out,          // [B,T]
                       unsigned* __restrict__ bar) {
    const int lane   = threadIdx.x & 31;
    const int wave   = threadIdx.x >> 5;
    const int gw     = blockIdx.x * WAVES_PER_BLOCK + wave; // 0..63
    const int m      = gw >> 4;                  // batch tile (0..3), rows m*16..
    const int hc     = gw & 15;                  // hidden-col tile (0..15)
    const int col    = lane & 15;                // N within tile
    const int rowoff = (lane >> 4) << 3;         // 0 or 8 (C/D row striping)
    const int ksel   = (lane >> 4) << 3;         // 0 or 8 (A/B K striping)
    const int arow   = m * 16 + (lane & 15);     // A-fragment source row

    // Per-lane gate constants for our 4 gate columns (i,f,g,o at +0,H,2H,3H)
    float wi[4], bb0[4], bb1[4];
#pragma unroll
    for (int g = 0; g < 4; ++g) {
        int gc = g * Hn + hc * 16 + col;
        wi[g]  = Wih0[gc];
        bb0[g] = b0s[gc];
        bb1[g] = b1s[gc];
    }
    const float fw = fcw[hc * 16 + col];

    v8f c0 = {0.f,0.f,0.f,0.f,0.f,0.f,0.f,0.f};
    v8f c1 = {0.f,0.f,0.f,0.f,0.f,0.f,0.f,0.f};

    for (int t = 0; t <= Tn; ++t) {
        // ---------- layer 0 at time t: gates = h0[t-1] @ Whh0^T ----------
        if (t < Tn) {
            const __bf16* h0old = h0buf + ((t + 1) & 1) * (Bn * Hn); // h0[t-1]
            __bf16*       h0new = h0buf + (t & 1) * (Bn * Hn);       // h0[t]
            const __bf16* abase = h0old + (size_t)arow * Hn + ksel;

            v8f acc[4];
#pragma unroll
            for (int g = 0; g < 4; ++g) acc[g] = (v8f){0.f,0.f,0.f,0.f,0.f,0.f,0.f,0.f};

            v16bf acur = load_frag(abase);
#pragma unroll
            for (int kc = 0; kc < Hn / 32; ++kc) {
                v16bf anext = acur;
                if (kc < Hn / 32 - 1) anext = load_frag(abase + (kc + 1) * 32);
#pragma unroll
                for (int g = 0; g < 4; ++g) {
                    v16bf bw = load_frag(whh0b + (size_t)(g * Hn + hc * 16 + col) * Hn
                                         + kc * 32 + ksel);
                    acc[g] = __builtin_amdgcn_wmma_f32_16x16x32_bf16(
                        false, acur, false, bw, (short)0, acc[g], false, false);
                }
                acur = anext;
            }

#pragma unroll
            for (int j = 0; j < 8; ++j) {
                const int row = m * 16 + rowoff + j;
                const float xv = x[(size_t)row * Tn + t];
                const float iv = sigmoidf_(acc[0][j] + xv * wi[0] + bb0[0]);
                const float fv = sigmoidf_(acc[1][j] + xv * wi[1] + bb0[1]);
                const float gv = tanhf_  (acc[2][j] + xv * wi[2] + bb0[2]);
                const float ov = sigmoidf_(acc[3][j] + xv * wi[3] + bb0[3]);
                const float cc = fv * c0[j] + iv * gv;
                c0[j] = cc;
                h0new[(size_t)row * Hn + hc * 16 + col] = (__bf16)(ov * tanhf_(cc));
            }
        }

        // ---------- layer 1 at time u=t-1: gates = [h0[u] | h1[u-1]] @ W1cat^T ----------
        if (t > 0) {
            const int u = t - 1;
            const __bf16* h0u   = h0buf + (u & 1) * (Bn * Hn);       // h0[u] (published)
            const __bf16* h1old = h1buf + ((u + 1) & 1) * (Bn * Hn); // h1[u-1]
            __bf16*       h1new = h1buf + (u & 1) * (Bn * Hn);       // h1[u]

            v8f acc[4];
#pragma unroll
            for (int g = 0; g < 4; ++g) acc[g] = (v8f){0.f,0.f,0.f,0.f,0.f,0.f,0.f,0.f};

            const __bf16* a0base = h0u   + (size_t)arow * Hn + ksel;
            const __bf16* a1base = h1old + (size_t)arow * Hn + ksel;

            v16bf acur = load_frag(a0base);
#pragma unroll
            for (int kc = 0; kc < 16; ++kc) {
                v16bf anext = acur;
                if (kc < 15) {
                    const int kn = kc + 1;
                    anext = load_frag(((kn < 8) ? a0base : a1base) + (kn & 7) * 32);
                }
#pragma unroll
                for (int g = 0; g < 4; ++g) {
                    v16bf bw = load_frag(w1b + (size_t)(g * Hn + hc * 16 + col) * 512
                                         + kc * 32 + ksel);
                    acc[g] = __builtin_amdgcn_wmma_f32_16x16x32_bf16(
                        false, acur, false, bw, (short)0, acc[g], false, false);
                }
                acur = anext;
            }

#pragma unroll
            for (int j = 0; j < 8; ++j) {
                const int row = m * 16 + rowoff + j;
                const float iv = sigmoidf_(acc[0][j] + bb1[0]);
                const float fv = sigmoidf_(acc[1][j] + bb1[1]);
                const float gv = tanhf_  (acc[2][j] + bb1[2]);
                const float ov = sigmoidf_(acc[3][j] + bb1[3]);
                const float cc = fv * c1[j] + iv * gv;
                c1[j] = cc;
                const float hv = ov * tanhf_(cc);
                h1new[(size_t)row * Hn + hc * 16 + col] = (__bf16)hv;

                // FC head: out[row,u] += sum over our 16 cols of hv*fc_w
                float pj = hv * fw;
                pj += __shfl_xor(pj, 1);
                pj += __shfl_xor(pj, 2);
                pj += __shfl_xor(pj, 4);
                pj += __shfl_xor(pj, 8);
                if (col == 0) atomicAdd(&out[(size_t)row * Tn + u], pj);
            }
        }

        // Publish h0[t] and h1[t-1] for the next iteration.
        if (t < Tn) grid_sync(bar, NBLOCKS);
    }
}

// ---------------- launch ----------------

extern "C" void kernel_launch(void* const* d_in, const int* in_sizes, int n_in,
                              void* d_out, int out_size, void* d_ws, size_t ws_size,
                              hipStream_t stream) {
    const float* x    = (const float*)d_in[0];
    const float* Wih0 = (const float*)d_in[1];
    const float* Whh0 = (const float*)d_in[2];
    const float* bih0 = (const float*)d_in[3];
    const float* bhh0 = (const float*)d_in[4];
    const float* Wih1 = (const float*)d_in[5];
    const float* Whh1 = (const float*)d_in[6];
    const float* bih1 = (const float*)d_in[7];
    const float* bhh1 = (const float*)d_in[8];
    const float* fc_w = (const float*)d_in[9];
    const float* fc_b = (const float*)d_in[10];
    float* out = (float*)d_out;

    char* w = (char*)d_ws;
    unsigned* bar   = (unsigned*)w;                                   // 256 B pad
    __bf16*   whh0b = (__bf16*)(w + 256);                             // 512 KB
    __bf16*   w1b   = (__bf16*)(w + 256 + 524288);                    // 1 MB
    float*    b0s   = (float*)(w + 256 + 524288 + 1048576);           // 4 KB
    float*    b1s   = b0s + 1024;                                     // 4 KB
    __bf16*   h0buf = (__bf16*)((char*)(b1s + 1024));                 // 64 KB
    __bf16*   h1buf = h0buf + 2 * Bn * Hn;                            // 64 KB

    lstm_init_kernel<<<256, 256, 0, stream>>>(Whh0, bih0, bhh0, Wih1, Whh1, bih1, bhh1,
                                              fc_b, out, whh0b, w1b, b0s, b1s,
                                              h0buf, h1buf, bar);

    lstm_persistent_kernel<<<NBLOCKS, NTHREADS, 0, stream>>>(
        x, Wih0, fc_w, whh0b, w1b, b0s, b1s, h0buf, h1buf, out, bar);
}
